// SingleHeadAttention_58987080843570
// MI455X (gfx1250) — matrix operand order repaired and checked by reading, not compiled, not run on hardware
//
#include <hip/hip_runtime.h>
#include <hip/hip_bf16.h>
#include <stdint.h>

// ---------------------------------------------------------------------------
// Single-head attention for MI455X (gfx1250, wave32, WMMA bf16 16x16x32).
// Pass 0: convert x and Wk/Wq/Wv to bf16 (Wq pre-scaled by 1/sqrt(C)=1/32).
// Pass 1: project xb -> K,Q,V in bf16 (pure bf16 WMMA GEMM; B-fragments
//         preloaded in batches so WMMAs overlap outstanding loads).
// Pass 2: flash attention, fp32 accumulation, online softmax.
// Workspace: 24 MB (K,Q,Vt) + 64 MB (xb) + 0.75 MB (Wb) in d_ws.
// ---------------------------------------------------------------------------

typedef __bf16 bf16_t;
typedef __bf16 v16bf __attribute__((ext_vector_type(16)));
typedef __bf16 v8bf  __attribute__((ext_vector_type(8)));
typedef float  v8f   __attribute__((ext_vector_type(8)));
typedef float  v4f   __attribute__((ext_vector_type(4)));

#define N_EMBD 1024
#define HEAD   128
#define SEQ    4096
#define BATCH  8

__device__ __forceinline__ v8f wmma_bf16(v16bf a, v16bf b, v8f c) {
  // 8 args: (neg_a, A, neg_b, B, c_mod, C, reuse_a, reuse_b)
  return __builtin_amdgcn_wmma_f32_16x16x32_bf16(false, a, false, b, (short)0, c,
                                                 false, false);
}

__device__ __forceinline__ v4f ld4(const float* p) { return *(const v4f*)p; }

// A-matrix fragment (16-bit, 16xK=32): per-lane chunks at p[0..7] and p[16..23].
__device__ __forceinline__ v16bf load_a16(const bf16_t* p) {
  v8bf lo = *(const v8bf*)p;
  v8bf hi = *(const v8bf*)(p + 16);
  return __builtin_shufflevector(lo, hi, 0, 1, 2, 3, 4, 5, 6, 7,
                                 8, 9, 10, 11, 12, 13, 14, 15);
}

// B-matrix fragment (16-bit, K=32 x 16): per-lane 16 contiguous K values.
__device__ __forceinline__ v16bf load_b16(const bf16_t* p) {
  v8bf lo = *(const v8bf*)p;
  v8bf hi = *(const v8bf*)(p + 8);
  return __builtin_shufflevector(lo, hi, 0, 1, 2, 3, 4, 5, 6, 7,
                                 8, 9, 10, 11, 12, 13, 14, 15);
}

// ---------------------------------------------------------------------------
// Pass 0: fp32 -> bf16 conversion (optionally scaled), 8 elements/thread.
// ---------------------------------------------------------------------------
__global__ void __launch_bounds__(256)
cvt_f32_bf16(const float* __restrict__ s, bf16_t* __restrict__ d, float sc,
             int n8) {
  const int i = blockIdx.x * blockDim.x + threadIdx.x;
  if (i >= n8) return;
  const float* p = s + (size_t)i * 8;
  v4f a = ld4(p), b = ld4(p + 4);
  v8bf r;
#pragma unroll
  for (int j = 0; j < 4; ++j) {
    r[j]     = (bf16_t)(a[j] * sc);
    r[4 + j] = (bf16_t)(b[j] * sc);
  }
  *(v8bf*)(d + (size_t)i * 8) = r;
}

// ---------------------------------------------------------------------------
// Pass 1: K/Q/V projection GEMM (all-bf16).  grid=(BT/128, 3), block=256.
// Wave w computes rows [blockIdx.x*128 + w*16, +16) x all 128 head cols of one
// projection (blockIdx.y selects which converted weight / output).
// ---------------------------------------------------------------------------
__global__ void __launch_bounds__(256)
proj_kqv(const bf16_t* __restrict__ xb, const bf16_t* __restrict__ wb,
         bf16_t* __restrict__ kb, bf16_t* __restrict__ qb,
         bf16_t* __restrict__ vT) {
  const int tid  = threadIdx.x;
  const int w    = tid >> 5;
  const int lane = tid & 31;
  const int hl   = lane >> 4;   // half-wave select
  const int l    = lane & 15;
  const int m0   = blockIdx.x * 128 + w * 16;
  const int y    = blockIdx.y;
  const bf16_t* W = wb + (size_t)y * HEAD * N_EMBD;

  v8f acc[8] = {};

  const bf16_t* xr0 = xb + (size_t)(m0 + l) * N_EMBD + hl * 16;
  const bf16_t* wr0 = W + (size_t)l * N_EMBD + hl * 16;
  for (int k0 = 0; k0 < N_EMBD; k0 += 32) {
    // A fragment: row (m0+l), K = hl*16+0..15
    v16bf a = load_b16(xr0 + k0);
    // Preload ALL 8 B-fragments (distinct registers -> loads batch into
    // clauses; WMMAs below overlap remaining outstanding loads)
    v16bf bf[8];
#pragma unroll
    for (int n = 0; n < 8; ++n)
      bf[n] = load_b16(wr0 + (size_t)(n * 16) * N_EMBD + k0);
#pragma unroll
    for (int n = 0; n < 8; ++n)
      acc[n] = wmma_bf16(a, bf[n], acc[n]);
  }

  // C/D layout: VGPR j, lanes 0-15 -> row j; lanes 16-31 -> row j+8; col = n*16+l.
  if (y < 2) {
    bf16_t* dst  = (y == 0) ? kb : qb;   // Wq already pre-scaled by 1/32
    bf16_t* drow = dst + (size_t)(m0 + hl * 8) * HEAD + l;
#pragma unroll
    for (int n = 0; n < 8; ++n)
#pragma unroll
      for (int j = 0; j < 8; ++j)
        drow[(size_t)j * HEAD + n * 16] = (bf16_t)acc[n][j];
  } else {
    // V stored transposed: vT[b*HEAD + h][t], packed pair stores (t, t+1).
    const int bb = m0 >> 12;
    const int tb = (m0 & (SEQ - 1)) + hl * 8;
#pragma unroll
    for (int n = 0; n < 8; ++n) {
      unsigned* dp = (unsigned*)(vT + (size_t)(bb * HEAD + n * 16 + l) * SEQ + tb);
#pragma unroll
      for (int j = 0; j < 8; j += 2) {
        unsigned p0 = (unsigned)__builtin_bit_cast(unsigned short, (bf16_t)acc[n][j]);
        unsigned p1 = (unsigned)__builtin_bit_cast(unsigned short, (bf16_t)acc[n][j + 1]);
        dp[j >> 1] = p0 | (p1 << 16);
      }
    }
  }
}

// ---------------------------------------------------------------------------
// Pass 2: flash attention.  grid=(SEQ/64, BATCH), block=128 (4 waves).
// Each wave owns 16 query rows; iterates key tiles of 32.
// K/V for all batches = 16 MB bf16 -> L2-resident across the sweep.
// ---------------------------------------------------------------------------
__global__ void __launch_bounds__(128)
flash_attn(const bf16_t* __restrict__ kb, const bf16_t* __restrict__ qb,
           const bf16_t* __restrict__ vT, float* __restrict__ out) {
  __shared__ __align__(16) bf16_t pbuf[4][16 * 32];  // exp(S) staging per wave
  const int tid  = threadIdx.x;
  const int w    = tid >> 5;
  const int lane = tid & 31;
  const int hl   = lane >> 4;
  const int l    = lane & 15;
  const int b    = blockIdx.y;
  const int q0   = blockIdx.x * 64 + w * 16;
  bf16_t* pb = pbuf[w];

  // Q fragments (contiguous K), kept in registers for the whole loop (H=128 -> 4 frags)
  const bf16_t* qrow = qb + (size_t)(b * SEQ + q0 + l) * HEAD + hl * 16;
  v16bf qf[4];
#pragma unroll
  for (int i = 0; i < 4; ++i) qf[i] = load_b16(qrow + i * 32);

  v8f o[8] = {};
  float mrow[8], lrow[8];
#pragma unroll
  for (int j = 0; j < 8; ++j) { mrow[j] = -3.0e38f; lrow[j] = 0.0f; }

  const bf16_t* kbase = kb + (size_t)b * SEQ * HEAD;
  const bf16_t* vbase = vT + (size_t)b * HEAD * SEQ;

  for (int kt = 0; kt < SEQ; kt += 32) {
    if (kt + 32 < SEQ) {
      __builtin_prefetch(kbase + (size_t)(kt + 32 + l) * HEAD, 0, 1);
      __builtin_prefetch(vbase + (size_t)(l * 16) * SEQ + kt + 32, 0, 1);
    }

    // --- S = Q * K^T  (16 x 32, fp32 acc); preload all 8 K-fragments ---
    const bf16_t* kp0 = kbase + (size_t)(kt + l) * HEAD + hl * 16;
    const bf16_t* kp1 = kp0 + 16 * HEAD;
    v16bf kf[8];
#pragma unroll
    for (int i = 0; i < 4; ++i) {
      kf[i]     = load_b16(kp0 + i * 32);
      kf[4 + i] = load_b16(kp1 + i * 32);
    }
    v8f s0 = {}, s1 = {};
#pragma unroll
    for (int i = 0; i < 4; ++i) {
      s0 = wmma_bf16(qf[i], kf[i], s0);
      s1 = wmma_bf16(qf[i], kf[4 + i], s1);
    }

    // --- online softmax (row j+hl*8 lives in VGPR j across one 16-lane half) ---
    float corr[8];
#pragma unroll
    for (int j = 0; j < 8; ++j) {
      float pm = fmaxf(s0[j], s1[j]);
      pm = fmaxf(pm, __shfl_xor(pm, 1));
      pm = fmaxf(pm, __shfl_xor(pm, 2));
      pm = fmaxf(pm, __shfl_xor(pm, 4));
      pm = fmaxf(pm, __shfl_xor(pm, 8));
      float mn = fmaxf(mrow[j], pm);
      float c  = __expf(mrow[j] - mn);
      float e0 = __expf(s0[j] - mn);
      float e1 = __expf(s1[j] - mn);
      float rs = e0 + e1;
      rs += __shfl_xor(rs, 1);
      rs += __shfl_xor(rs, 2);
      rs += __shfl_xor(rs, 4);
      rs += __shfl_xor(rs, 8);
      lrow[j] = lrow[j] * c + rs;
      mrow[j] = mn;
      corr[j] = c;
      // stage exp(S) into LDS in row-major [16 rows][32 keys] (bf16)
      pb[(j + hl * 8) * 32 + l]      = (bf16_t)e0;
      pb[(j + hl * 8) * 32 + 16 + l] = (bf16_t)e1;
    }

    // --- O += P * V: preload all 8 V-fragments while applying correction ---
    v16bf vf[8];
#pragma unroll
    for (int n = 0; n < 8; ++n)
      vf[n] = load_b16(vbase + (size_t)(n * 16 + l) * SEQ + kt + hl * 16);

#pragma unroll
    for (int n = 0; n < 8; ++n)
#pragma unroll
      for (int j = 0; j < 8; ++j) o[n][j] *= corr[j];

    // wait for same-wave LDS stores, then reload P in A-fragment layout
    asm volatile("s_wait_dscnt 0" ::: "memory");
    v16bf pf = load_b16(pb + l * 32 + hl * 16);

#pragma unroll
    for (int n = 0; n < 8; ++n)
      o[n] = wmma_bf16(pf, vf[n], o[n]);
  }

  // epilogue: normalize and store fp32 output
  float inv[8];
#pragma unroll
  for (int j = 0; j < 8; ++j) inv[j] = 1.0f / lrow[j];
  float* orow = out + (size_t)(b * SEQ + q0 + hl * 8) * HEAD;
#pragma unroll
  for (int n = 0; n < 8; ++n)
#pragma unroll
    for (int j = 0; j < 8; ++j)
      orow[(size_t)j * HEAD + n * 16 + l] = o[n][j] * inv[j];
}

// ---------------------------------------------------------------------------
extern "C" void kernel_launch(void* const* d_in, const int* in_sizes, int n_in,
                              void* d_out, int out_size, void* d_ws, size_t ws_size,
                              hipStream_t stream) {
  const float* x  = (const float*)d_in[0];
  const float* Wk = (const float*)d_in[1];
  const float* Wq = (const float*)d_in[2];
  const float* Wv = (const float*)d_in[3];

  const size_t BT = (size_t)BATCH * SEQ;          // 32768
  bf16_t* kb = (bf16_t*)d_ws;                     // [BT][HEAD]
  bf16_t* qb = kb + BT * HEAD;                    // [BT][HEAD] (pre-scaled)
  bf16_t* vT = qb + BT * HEAD;                    // [BATCH*HEAD][SEQ]
  bf16_t* xb = vT + BT * HEAD;                    // [BT][N_EMBD]
  bf16_t* wb = xb + BT * N_EMBD;                  // [3][HEAD][N_EMBD]

  // Pass 0: bf16 conversion (fold 1/sqrt(C) into Wq)
  const int xn8 = (int)(BT * N_EMBD / 8);
  const int wn8 = HEAD * N_EMBD / 8;
  cvt_f32_bf16<<<xn8 / 256, 256, 0, stream>>>(x, xb, 1.0f, xn8);
  cvt_f32_bf16<<<wn8 / 256, 256, 0, stream>>>(Wk, wb, 1.0f, wn8);
  cvt_f32_bf16<<<wn8 / 256, 256, 0, stream>>>(Wq, wb + (size_t)1 * HEAD * N_EMBD,
                                              0.03125f, wn8);
  cvt_f32_bf16<<<wn8 / 256, 256, 0, stream>>>(Wv, wb + (size_t)2 * HEAD * N_EMBD,
                                              1.0f, wn8);

  proj_kqv<<<dim3((unsigned)(BT / 128), 3), 256, 0, stream>>>(xb, wb, kb, qb, vT);
  flash_attn<<<dim3(SEQ / 64, BATCH), 128, 0, stream>>>(kb, qb, vT,
                                                        (float*)d_out);
}
